// Refinement_16896401342899
// MI455X (gfx1250) — compile-verified, matrix operand order checked by reading, not run on hardware
//
#include <hip/hip_runtime.h>
#include <hip/hip_bf16.h>

// ---------- types ----------
typedef __attribute__((ext_vector_type(16))) __bf16 v16bf;
typedef __attribute__((ext_vector_type(8)))  __bf16 v8bf;
typedef __attribute__((ext_vector_type(8)))  float  v8f;
typedef __attribute__((ext_vector_type(4)))  unsigned int v4u;
typedef __attribute__((ext_vector_type(8)))  int    v8i;
typedef __attribute__((ext_vector_type(4)))  int    v4i;

#define N_IMG 32
#define N_CAP 32
#define N_RGN 36
#define MAX_W 50
#define EMBED 1024

// Tensor Data Mover path (6-arg builtin on this toolchain); falls back to direct
// global loads if the builtin is unavailable (also keeps the host pass legal).
#if defined(__gfx1250__) && __has_builtin(__builtin_amdgcn_tensor_load_to_lds) && \
    __has_builtin(__builtin_amdgcn_s_wait_tensorcnt)
#define USE_TDM 1
#else
#define USE_TDM 0
#endif

static __device__ __forceinline__ __bf16 f2bf(float f) {
    unsigned u = __builtin_bit_cast(unsigned, f);
    unsigned lsb = (u >> 16) & 1u;
    u += 0x7fffu + lsb;
    unsigned short h = (unsigned short)(u >> 16);
    return __builtin_bit_cast(__bf16, h);
}

static __device__ __forceinline__ v8f vzero8() {
    v8f z = {0.f, 0.f, 0.f, 0.f, 0.f, 0.f, 0.f, 0.f};
    return z;
}

// Load a 16-element bf16 WMMA A/B fragment: two contiguous 16B chunks,
// p must already point at (row_base + k_base + h*8).
static __device__ __forceinline__ v16bf load_frag(const __bf16* p) {
    v8bf lo = *(const v8bf*)(p);
    v8bf hi = *(const v8bf*)(p + 16);
    v16bf r;
#pragma unroll
    for (int e = 0; e < 8; ++e) { r[e] = lo[e]; r[e + 8] = hi[e]; }
    return r;
}

static __device__ __forceinline__ v8f wmma_bf16(v16bf a, v16bf b, v8f c) {
    return __builtin_amdgcn_wmma_f32_16x16x32_bf16(false, a, false, b, (short)0, c,
                                                   false, false);
}

// ---------- K0: fp32 -> bf16 conversion ----------
__global__ __launch_bounds__(256) void cvt_bf16_kernel(const float* __restrict__ src,
                                                       __bf16* __restrict__ dst, int n) {
    int idx = blockIdx.x * 256 + threadIdx.x;
    if (idx < n) dst[idx] = f2bf(src[idx]);
}

// wrdT[c][d][l(64, zero padded)]  <- wrd[c][l][d]
__global__ __launch_bounds__(256) void build_wrdT_kernel(const float* __restrict__ wrd,
                                                         __bf16* __restrict__ wrdT) {
    int idx = blockIdx.x * 256 + threadIdx.x;   // C*D*64 = 2^21
    int l = idx & 63;
    int d = (idx >> 6) & (EMBED - 1);
    int c = idx >> 16;
    float v = (l < MAX_W) ? wrd[((size_t)c * MAX_W + l) * EMBED + d] : 0.f;
    wrdT[idx] = f2bf(v);
}

// ---------- K1: scores -> softmax attention -> weighted context (bf16) ----------
// grid (I, C), 128 threads (4 waves).
__global__ __launch_bounds__(128) void attn_wei_kernel(
    const __bf16* __restrict__ wrdb,   // C x L x D
    const __bf16* __restrict__ rgnb,   // I x Q x D
    const __bf16* __restrict__ wrdT,   // C x D x 64 (zero-padded l)
    const int*    __restrict__ lens,   // C
    __bf16*       __restrict__ wei)    // C x I x Q x D
{
    const int i   = blockIdx.x;
    const int c   = blockIdx.y;
    const int tid = threadIdx.x;
    const int lane = tid & 31, wv = tid >> 5;
    const int ln16 = lane & 15, hh = lane >> 4;

    __shared__ float  S[64][49];       // scores, rows l (pad 50->64), cols q (pad 36->48)
    __shared__ __bf16 aT[48][88];      // attn^T (q rows, l cols), 16B-aligned pitch
    __shared__ float  stg[4][48 * 16]; // per-wave store-coalescing staging
#if USE_TDM
    // wrdT slab for this caption, DMA'd by the TDM with 16B padding per 128B row:
    // row pitch = 72 bf16 = 144B = 36 dwords -> 16B aligned, bank-conflict-free
    // for the 16-lane fragment reads (36*l mod 64 distinct for l in 0..15).
    __shared__ alignas(16) __bf16 lwrd[EMBED * 72];   // 144 KB

    // ---- issue the TDM load early: overlaps the whole 128KB fetch with step 1/2 ----
    if (wv == 0) {
        unsigned lds_off = (unsigned)(size_t)(void*)&lwrd[0];
        unsigned long long ga =
            (unsigned long long)(size_t)(wrdT + (size_t)c * (EMBED * 64));
        // D# group 0: count=1 | lds_addr | global_addr[56:0] | type=2
        v4u g0 = {1u, lds_off, (unsigned)ga,
                  (unsigned)((ga >> 32) & 0x1FFFFFFu) | 0x80000000u};
        // D# group 1: data_size=2B(code1), pad_enable, pad_interval=32dw(code4),
        // pad_amount=4dw(code3); tensor 64 x 1024, tile 64 x 1024, dim0 stride 64,
        // dim1 stride 65536 (unused for 2D tile but kept consistent).
        v8i g1 = {(int)((1u << 16) | (1u << 20) | (4u << 22) | (3u << 25)),
                  (int)(64u << 16),    // tensor_dim0 = 64
                  (int)(1024u << 16),  // tensor_dim1 = 1024
                  (int)(64u << 16),    // tile_dim0 = 64
                  (int)1024,           // tile_dim1 = 1024, tile_dim2 = 0
                  (int)64,             // tensor_dim0_stride = 64
                  0,                   // dim0_stride hi / dim1_stride lo
                  1};                  // tensor_dim1_stride = 65536
        v4i gz4 = {0, 0, 0, 0};
        v8i gz8 = {0, 0, 0, 0, 0, 0, 0, 0};
        __builtin_amdgcn_tensor_load_to_lds(g0, g1, gz4, gz4, gz8, 0);
    }
#endif

    // ---- step 1: S[l][q] = wrd_c[l] . rgn_i[q] via WMMA (K over D) ----
    {
        int l = 16 * wv + ln16; if (l > MAX_W - 1) l = MAX_W - 1;  // clamp pad rows
        const __bf16* aBase = wrdb + ((size_t)c * MAX_W + l) * EMBED + hh * 8;
        const __bf16* bBase[3];
#pragma unroll
        for (int qt = 0; qt < 3; ++qt) {
            int q = qt * 16 + ln16; if (q > N_RGN - 1) q = N_RGN - 1;
            bBase[qt] = rgnb + ((size_t)i * N_RGN + q) * EMBED + hh * 8;
        }
        v8f acc[3] = {vzero8(), vzero8(), vzero8()};
        for (int kt = 0; kt < EMBED / 32; ++kt) {
            v16bf a = load_frag(aBase + kt * 32);
#pragma unroll
            for (int qt = 0; qt < 3; ++qt) {
                v16bf b = load_frag(bBase[qt] + kt * 32);
                acc[qt] = wmma_bf16(a, b, acc[qt]);
            }
        }
#pragma unroll
        for (int qt = 0; qt < 3; ++qt)
#pragma unroll
            for (int v = 0; v < 8; ++v)
                S[16 * wv + v + 8 * hh][qt * 16 + ln16] = acc[qt][v];
    }
    __syncthreads();

    // ---- step 2a: per-row LeakyReLU + L2 normalize over q, scale by lambda ----
    if (tid < MAX_W) {
        float ss = 0.f;
        for (int q = 0; q < N_RGN; ++q) {
            float x = S[tid][q];
            x = (x >= 0.f) ? x : 0.1f * x;
            ss += x * x;
        }
        float inv = 9.0f / fmaxf(sqrtf(ss), 1e-12f);
        for (int q = 0; q < N_RGN; ++q) {
            float x = S[tid][q];
            x = (x >= 0.f) ? x : 0.1f * x;
            S[tid][q] = x * inv;
        }
    }
    __syncthreads();

    // ---- step 2b: masked softmax over l (per column q), write attn^T bf16 ----
    if (tid < 48) {
        const int q   = tid;
        const int len = lens[c];
        if (q < N_RGN) {
            float m = -1e30f;
            for (int l = 0; l < len; ++l) m = fmaxf(m, S[l][q]);
            float sum = 0.f;
            for (int l = 0; l < len; ++l) sum += __expf(S[l][q] - m);
            float inv = 1.0f / sum;
            for (int l = 0; l < 64; ++l)
                aT[q][l] = (l < len) ? f2bf(__expf(S[l][q] - m) * inv) : f2bf(0.f);
        } else {
            for (int l = 0; l < 64; ++l) aT[q][l] = f2bf(0.f);
        }
    }
#if USE_TDM
    if (wv == 0) __builtin_amdgcn_s_wait_tensorcnt(0);  // TDM data in LDS
#endif
    __syncthreads();

    // ---- step 3: wei[q][d] = attn^T (48x64) @ wrd_c (64xD) via WMMA ----
    v16bf afr[3][2];
#pragma unroll
    for (int mt = 0; mt < 3; ++mt)
#pragma unroll
        for (int kt = 0; kt < 2; ++kt)
            afr[mt][kt] = load_frag(&aT[mt * 16 + ln16][kt * 32 + hh * 8]);

    for (int nt = wv; nt < EMBED / 16; nt += 4) {
        v8f acc[3] = {vzero8(), vzero8(), vzero8()};
#pragma unroll
        for (int kt = 0; kt < 2; ++kt) {
#if USE_TDM
            const __bf16* bp =
                &lwrd[(size_t)(nt * 16 + ln16) * 72 + kt * 32 + hh * 8];
#else
            const __bf16* bp =
                wrdT + ((size_t)c * EMBED + nt * 16 + ln16) * 64 + kt * 32 + hh * 8;
#endif
            v16bf b = load_frag(bp);
#pragma unroll
            for (int mt = 0; mt < 3; ++mt) acc[mt] = wmma_bf16(afr[mt][kt], b, acc[mt]);
        }
        // stage 48x16 tile in LDS (per-wave region; LDS ops are in-order per wave)
#pragma unroll
        for (int mt = 0; mt < 3; ++mt)
#pragma unroll
            for (int v = 0; v < 8; ++v)
                stg[wv][(mt * 16 + v + 8 * hh) * 16 + ln16] = acc[mt][v];
        // coalesced-ish row stores: each lane writes one 32B row chunk
        for (int rr = lane; rr < N_RGN; rr += 32) {
            v8bf o0, o1;
#pragma unroll
            for (int j = 0; j < 8; ++j) {
                o0[j] = f2bf(stg[wv][rr * 16 + j]);
                o1[j] = f2bf(stg[wv][rr * 16 + 8 + j]);
            }
            __bf16* dst =
                wei + (((size_t)c * N_IMG + i) * N_RGN + rr) * EMBED + nt * 16;
            *(v8bf*)dst       = o0;
            *(v8bf*)(dst + 8) = o1;
        }
    }
}

// ---------- K2/K3/K4: fused FiLM GEMMs ----------
// M = C*I*Q = 36864 rows, N = K = 1024.  Block tile 64x64, 4 waves x (32x32).
// MODE 0: t0 = A@Wscale^T, t1 = A@Wshift^T; x = rgn*tanh(t0+b0) + (t1+b1) -> bf16
// MODE 1: h = relu(A@W1^T + b0) -> bf16
// MODE 2: out[i,c,q,:] = A@W2^T + b0 + rgn[i,q,:] -> fp32 (with (c,i) swap)
template <int MODE>
__global__ __launch_bounds__(128) void film_gemm_kernel(
    const __bf16* __restrict__ A,
    const __bf16* __restrict__ B0w,
    const __bf16* __restrict__ B1w,
    const float*  __restrict__ bias0,
    const float*  __restrict__ bias1,
    const float*  __restrict__ rgn,
    __bf16*       __restrict__ outb,
    float*        __restrict__ outf)
{
    const int tid = threadIdx.x;
    const int lane = tid & 31, wv = tid >> 5;
    const int ln16 = lane & 15, hh = lane >> 4;
    const int m0 = blockIdx.y * 64 + (wv >> 1) * 32;
    const int n0 = blockIdx.x * 64 + (wv & 1) * 32;

    v8f acc0[2][2], acc1[2][2];
#pragma unroll
    for (int mi = 0; mi < 2; ++mi)
#pragma unroll
        for (int ni = 0; ni < 2; ++ni) { acc0[mi][ni] = vzero8(); acc1[mi][ni] = vzero8(); }

    const __bf16* aBase[2];
    const __bf16* b0Base[2];
    const __bf16* b1Base[2];
#pragma unroll
    for (int mi = 0; mi < 2; ++mi)
        aBase[mi] = A + (size_t)(m0 + mi * 16 + ln16) * EMBED + hh * 8;
#pragma unroll
    for (int ni = 0; ni < 2; ++ni) {
        b0Base[ni] = B0w + (size_t)(n0 + ni * 16 + ln16) * EMBED + hh * 8;
        b1Base[ni] = (MODE == 0)
                         ? (B1w + (size_t)(n0 + ni * 16 + ln16) * EMBED + hh * 8)
                         : b0Base[ni];
    }

    for (int kt = 0; kt < EMBED / 32; ++kt) {
        v16bf a[2], b0[2];
#pragma unroll
        for (int mi = 0; mi < 2; ++mi) a[mi] = load_frag(aBase[mi] + kt * 32);
#pragma unroll
        for (int ni = 0; ni < 2; ++ni) b0[ni] = load_frag(b0Base[ni] + kt * 32);
#pragma unroll
        for (int mi = 0; mi < 2; ++mi)
#pragma unroll
            for (int ni = 0; ni < 2; ++ni)
                acc0[mi][ni] = wmma_bf16(a[mi], b0[ni], acc0[mi][ni]);
        if (MODE == 0) {
            v16bf b1[2];
#pragma unroll
            for (int ni = 0; ni < 2; ++ni) b1[ni] = load_frag(b1Base[ni] + kt * 32);
#pragma unroll
            for (int mi = 0; mi < 2; ++mi)
#pragma unroll
                for (int ni = 0; ni < 2; ++ni)
                    acc1[mi][ni] = wmma_bf16(a[mi], b1[ni], acc1[mi][ni]);
        }
    }

    // epilogue
#pragma unroll
    for (int mi = 0; mi < 2; ++mi) {
#pragma unroll
        for (int ni = 0; ni < 2; ++ni) {
            const int n = n0 + ni * 16 + ln16;
            const float bb0 = bias0[n];
            const float bb1 = (MODE == 0) ? bias1[n] : 0.f;
#pragma unroll
            for (int v = 0; v < 8; ++v) {
                const int r = m0 + mi * 16 + v + 8 * hh;  // r = c*(I*Q) + i*Q + q
                if (MODE == 0) {
                    const int iq = r % (N_IMG * N_RGN);
                    float sc = tanhf(acc0[mi][ni][v] + bb0);
                    float sh = acc1[mi][ni][v] + bb1;
                    float rg = rgn[(size_t)iq * EMBED + n];
                    outb[(size_t)r * EMBED + n] = f2bf(rg * sc + sh);
                } else if (MODE == 1) {
                    float hv = acc0[mi][ni][v] + bb0;
                    outb[(size_t)r * EMBED + n] = f2bf(fmaxf(hv, 0.f));
                } else {
                    const int cc = r / (N_IMG * N_RGN);
                    const int iq = r % (N_IMG * N_RGN);
                    const int ii = iq / N_RGN;
                    const int qq = iq % N_RGN;
                    float rg = rgn[(size_t)iq * EMBED + n];
                    outf[(((size_t)ii * N_CAP + cc) * N_RGN + qq) * EMBED + n] =
                        acc0[mi][ni][v] + bb0 + rg;
                }
            }
        }
    }
}

// ---------- host launcher ----------
extern "C" void kernel_launch(void* const* d_in, const int* in_sizes, int n_in,
                              void* d_out, int out_size, void* d_ws, size_t ws_size,
                              hipStream_t stream) {
    (void)in_sizes; (void)n_in; (void)out_size; (void)ws_size;
    const float* rgn     = (const float*)d_in[0];
    const float* wrd     = (const float*)d_in[2];
    const int*   lens    = (const int*)d_in[4];
    const float* w_scale = (const float*)d_in[5];
    const float* b_scale = (const float*)d_in[6];
    const float* w_shift = (const float*)d_in[7];
    const float* b_shift = (const float*)d_in[8];
    const float* w1      = (const float*)d_in[9];
    const float* b1      = (const float*)d_in[10];
    const float* w2      = (const float*)d_in[11];
    const float* b2      = (const float*)d_in[12];

    char* ws = (char*)d_ws;
    const size_t SZ_WRDB = (size_t)N_CAP * MAX_W * EMBED * 2;   // 3,276,800
    const size_t SZ_RGNB = (size_t)N_IMG * N_RGN * EMBED * 2;   // 2,359,296
    const size_t SZ_WRDT = (size_t)N_CAP * EMBED * 64 * 2;      // 4,194,304
    const size_t SZ_W    = (size_t)EMBED * EMBED * 2;           // 2,097,152
    const size_t SZ_BIG  = (size_t)N_CAP * N_IMG * N_RGN * EMBED * 2;  // 75,497,472

    __bf16* wrdb = (__bf16*)(ws);
    __bf16* rgnb = (__bf16*)(ws + SZ_WRDB);
    __bf16* wrdT = (__bf16*)(ws + SZ_WRDB + SZ_RGNB);
    __bf16* wscb = (__bf16*)(ws + SZ_WRDB + SZ_RGNB + SZ_WRDT);
    __bf16* wshb = (__bf16*)((char*)wscb + SZ_W);
    __bf16* w1b  = (__bf16*)((char*)wscb + 2 * SZ_W);
    __bf16* w2b  = (__bf16*)((char*)wscb + 3 * SZ_W);
    __bf16* wei  = (__bf16*)((char*)wscb + 4 * SZ_W);
    __bf16* xb   = (__bf16*)((char*)wei + SZ_BIG);
    __bf16* hb   = wei;  // wei is dead after mode-0 GEMM; reuse its buffer

    // K0: conversions
    {
        int n;
        n = N_CAP * MAX_W * EMBED;
        cvt_bf16_kernel<<<(n + 255) / 256, 256, 0, stream>>>(wrd, wrdb, n);
        n = N_IMG * N_RGN * EMBED;
        cvt_bf16_kernel<<<(n + 255) / 256, 256, 0, stream>>>(rgn, rgnb, n);
        n = EMBED * EMBED;
        cvt_bf16_kernel<<<(n + 255) / 256, 256, 0, stream>>>(w_scale, wscb, n);
        cvt_bf16_kernel<<<(n + 255) / 256, 256, 0, stream>>>(w_shift, wshb, n);
        cvt_bf16_kernel<<<(n + 255) / 256, 256, 0, stream>>>(w1, w1b, n);
        cvt_bf16_kernel<<<(n + 255) / 256, 256, 0, stream>>>(w2, w2b, n);
        int nt = N_CAP * EMBED * 64;
        build_wrdT_kernel<<<nt / 256, 256, 0, stream>>>(wrd, wrdT);
    }

    // K1: attention + weighted context (TDM-staged wrdT slab when available)
    attn_wei_kernel<<<dim3(N_IMG, N_CAP), 128, 0, stream>>>(wrdb, rgnb, wrdT, lens, wei);

    // K2..K4: FiLM GEMM chain
    const dim3 gemmGrid(EMBED / 64, (N_CAP * N_IMG * N_RGN) / 64);  // 16 x 576
    film_gemm_kernel<0><<<gemmGrid, 128, 0, stream>>>(wei, wscb, wshb, b_scale, b_shift,
                                                      rgn, xb, nullptr);
    film_gemm_kernel<1><<<gemmGrid, 128, 0, stream>>>(xb, w1b, nullptr, b1, nullptr,
                                                      nullptr, hb, nullptr);
    film_gemm_kernel<2><<<gemmGrid, 128, 0, stream>>>(hb, w2b, nullptr, b2, nullptr,
                                                      rgn, nullptr, (float*)d_out);
}
